// SlinkyForcePredictorCartesian_18820546691819
// MI455X (gfx1250) — matrix operand order repaired and checked by reading, not compiled, not run on hardware
//
#include <hip/hip_runtime.h>
#include <math.h>

typedef __attribute__((ext_vector_type(16))) _Float16 v16h;
typedef __attribute__((ext_vector_type(8)))  float    v8f;
typedef __attribute__((ext_vector_type(4)))  float    v4f;

#define N_NODES   30000
#define N_EDGES   480000
#define N_GRAPHS  10
#define NPG       3000

union alignas(32) HVec {
  v16h v;
  v4f  f[2];
  _Float16 h[16];
};

// ---------------------------------------------------------------------------
// Weight packing: store B-matrix (KxN, f32 row-major) in the per-lane v16h
// layout required by V_WMMA_F32_16X16X32_F16 (ISA 7.12.2):
//   lane<16 : N = lane,    half j <-> K = kc*32 + (j<8 ? j : 16 + j-8)
//   lane>=16: N = lane-16, half j <-> K = kc*32 + (j<8 ? 8 + j : 24 + j-8)
// dst index = ((nt*kchunks + kc)*32 + lane)*16 + j
// ---------------------------------------------------------------------------
__global__ void pack_b_kernel(const float* __restrict__ src, _Float16* __restrict__ dst,
                              int K, int N, int ntiles, int kchunks) {
  int t = blockIdx.x * blockDim.x + threadIdx.x;
  int total = ntiles * kchunks * 32 * 16;
  if (t >= total) return;
  int j    = t & 15;
  int lane = (t >> 4) & 31;
  int rest = t >> 9;
  int kc   = rest % kchunks;
  int nt   = rest / kchunks;
  int n = nt * 16 + (lane & 15);
  int k = kc * 32 + ((lane & 16) ? 8 : 0) + ((j < 8) ? j : 16 + (j - 8));
  float v = (k < K && n < N) ? src[(size_t)k * N + n] : 0.f;
  dst[t] = (_Float16)v;
}

__device__ __forceinline__ v16h load_pack(const _Float16* __restrict__ p, int idx, int lane) {
  HVec u;
  const v4f* q = (const v4f*)(p + ((size_t)idx * 32 + lane) * 16);
  u.f[0] = q[0];
  u.f[1] = q[1];
  return u.v;
}

// ---------------------------------------------------------------------------
// Edge preprocessing: ea[E][10] = [edge_attr, SH(lmax=2)], emb_h[E][16] (f16, padded)
// ---------------------------------------------------------------------------
__global__ void edge_prep_kernel(const float* __restrict__ pos,
                                 const float* __restrict__ edge_attr,
                                 const int* __restrict__ esrc,
                                 const int* __restrict__ edst,
                                 float* __restrict__ ea,
                                 _Float16* __restrict__ emb_h) {
  int e = blockIdx.x * blockDim.x + threadIdx.x;
  if (e >= N_EDGES) return;
  int s = esrc[e], d = edst[e];
  float vx = pos[s * 3 + 0] - pos[d * 3 + 0];
  float vy = pos[s * 3 + 1] - pos[d * 3 + 1];
  float vz = pos[s * 3 + 2] - pos[d * 3 + 2];
  float len = sqrtf(vx * vx + vy * vy + vz * vz);
  float inv = 1.f / (len + 1e-12f);
  float x = vx * inv, y = vy * inv, z = vz * inv;
  const float s3 = 1.7320508075688772f, s5 = 2.23606797749979f, s15 = 3.872983346207417f;
  float* o = ea + (size_t)e * 10;
  o[0] = edge_attr[e];
  o[1] = 1.f;
  o[2] = s3 * x;
  o[3] = s3 * y;
  o[4] = s3 * z;
  o[5] = s15 * x * z;
  o[6] = s15 * x * y;
  o[7] = s5 * (y * y - 0.5f * (x * x + z * z));
  o[8] = s15 * y * z;
  o[9] = 0.5f * s15 * (z * z - x * x);
  const float step = 4.f / 11.f;
  const float C = 1.14136f * 7.3890560989306495f;  // 1.14136 * e^2
  const float sq10 = 3.1622776601683795f;
  _Float16* eo = emb_h + (size_t)e * 16;
#pragma unroll
  for (int i = 0; i < 10; ++i) {
    float v = 4.f * (float)(i + 1) / 11.f;
    float dd = (len - v) / step;
    float d2 = dd * dd;
    float yv = (d2 < 1.f) ? C * __expf(-1.f / (1.f - d2)) * sq10 : 0.f;
    eo[i] = (_Float16)yv;
  }
#pragma unroll
  for (int i = 10; i < 16; ++i) eo[i] = (_Float16)0.f;
}

// ---------------------------------------------------------------------------
// Gate kernel: one wave per 16-edge tile; per layer l:
//   h = silu(emb @ fc1[l] + b1[l])   (WMMA 16x16x32, 7 column tiles of 100)
//   rad = h @ fc2[l]                 (LDS transpose + 4 K-chunk WMMAs)
//   g_all[l][e] = dot(ea[e], rad[e]) (half-wave shuffle reduction)
// ---------------------------------------------------------------------------
__global__ void gate_kernel(const _Float16* __restrict__ emb_h,
                            const float* __restrict__ ea,
                            const _Float16* __restrict__ fc1p,
                            const _Float16* __restrict__ fc2p,
                            const float* __restrict__ b1,
                            float* __restrict__ g_all) {
  __shared__ _Float16 hbuf[8][16][136];  // 272B row stride -> conflict-free b128 reads
  int wave = threadIdx.x >> 5;
  int lane = threadIdx.x & 31;
  int tile = blockIdx.x * 8 + wave;
  if (tile * 16 >= N_EDGES) return;

  // zero K = 112..135 tail once (second-GEMM reads K<128)
  if (lane < 16) {
#pragma unroll
    for (int j = 112; j < 136; ++j) hbuf[wave][lane][j] = (_Float16)0.f;
  }

  // A matrix: emb tile (16 edges x K32 padded from 10)
  HVec ua;
  {
    int e = tile * 16 + (lane & 15);
    const v4f* p = (const v4f*)(emb_h + (size_t)e * 16 + ((lane & 16) ? 8 : 0));
    ua.f[0] = p[0];
    v4f z = {0.f, 0.f, 0.f, 0.f};
    ua.f[1] = z;  // K = 16..31 (and 24..31) are zero
  }

  for (int l = 0; l < 4; ++l) {
    // ---- first MLP layer: 7 column tiles of 16 (covers 100 hidden + pad) ----
    for (int c = 0; c < 7; ++c) {
      v16h b = load_pack(fc1p, l * 7 + c, lane);
      v8f h = {};
      h = __builtin_amdgcn_wmma_f32_16x16x32_f16(false, ua.v, false, b, (short)0, h,
                                                 false, false);
      int col = c * 16 + (lane & 15);
      float bias = (col < 100) ? b1[l * 100 + col] : 0.f;
#pragma unroll
      for (int r = 0; r < 8; ++r) {
        float v = h[r] + bias;
        float sv = v / (1.f + __expf(-v));  // silu
        int m = (lane & 16) ? (8 + r) : r;
        hbuf[wave][m][col] = (_Float16)sv;
      }
    }
    // ---- second MLP layer: rad = h @ fc2, K=112 (4 chunks), N=16 (10 valid) ----
    v8f acc = {};
    for (int kk = 0; kk < 4; ++kk) {
      HVec ua2;
      const _Float16* hr = &hbuf[wave][lane & 15][0];
      ua2.f[0] = *(const v4f*)(hr + kk * 32 + ((lane & 16) ? 8 : 0));
      ua2.f[1] = *(const v4f*)(hr + kk * 32 + ((lane & 16) ? 24 : 16));
      v16h b = load_pack(fc2p, l * 4 + kk, lane);
      acc = __builtin_amdgcn_wmma_f32_16x16x32_f16(false, ua2.v, false, b, (short)0, acc,
                                                   false, false);
    }
    // ---- g = sum_n ea[e][n]*rad[e][n], reduce within each half-wave ----
    int n = lane & 15;
#pragma unroll
    for (int r = 0; r < 8; ++r) {
      int m = (lane & 16) ? (8 + r) : r;
      int e = tile * 16 + m;
      float p = (n < 10) ? acc[r] * ea[(size_t)e * 10 + n] : 0.f;
      p += __shfl_xor(p, 1, 16);
      p += __shfl_xor(p, 2, 16);
      p += __shfl_xor(p, 4, 16);
      p += __shfl_xor(p, 8, 16);
      if (n == 0) g_all[(size_t)l * N_EDGES + e] = p;
    }
  }
}

// ---------------------------------------------------------------------------
// x0 = node_input * node_attr (stored f16; all x activations live in f16,
// all accumulation stays f32)
// ---------------------------------------------------------------------------
__global__ void x0_kernel(const float* __restrict__ ni, const float* __restrict__ na,
                          _Float16* __restrict__ x) {
  int t = blockIdx.x * blockDim.x + threadIdx.x;
  if (t >= N_NODES * 8) return;
  x[t] = (_Float16)(ni[t] * na[t >> 3]);
}

__global__ void zero_kernel(float* __restrict__ p, int n) {
  int t = blockIdx.x * blockDim.x + threadIdx.x;
  if (t < n) p[t] = 0.f;
}

// scatter-add, 288-dim: one wave per edge iteration; f16 source rows (576B)
// halve the random-gather traffic; f32 atomics into L2-resident agg.
// Grid-stride with global_prefetch_b8 of the next edge's source row.
__global__ void scatter288_kernel(const _Float16* __restrict__ x, const float* __restrict__ g,
                                  const int* __restrict__ esrc, const int* __restrict__ edst,
                                  float* __restrict__ agg) {
  int w = (blockIdx.x * blockDim.x + threadIdx.x) >> 5;
  int lane = threadIdx.x & 31;
  int wtot = (gridDim.x * blockDim.x) >> 5;
  for (int e = w; e < N_EDGES; e += wtot) {
    int en = e + wtot;
    if (en < N_EDGES) {
      // one byte every 18B across 32 lanes touches every cacheline of the
      // next edge's 576B f16 source row (emits global_prefetch_b8)
      __builtin_prefetch(x + (size_t)esrc[en] * 288 + lane * 9, 0, 1);
    }
    int s = esrc[e], d = edst[e];
    float gv = g[e];
    const _Float16* xs = x + (size_t)s * 288;
    float* ad = agg + (size_t)d * 288;
#pragma unroll
    for (int i = 0; i < 9; ++i) {
      int c = lane + 32 * i;
      atomicAdd(&ad[c], (float)xs[c] * gv);
    }
  }
}

// scatter-add, 8-dim: 8 threads per edge
__global__ void scatter8_kernel(const _Float16* __restrict__ x, const float* __restrict__ g,
                                const int* __restrict__ esrc, const int* __restrict__ edst,
                                float* __restrict__ agg) {
  int t = blockIdx.x * blockDim.x + threadIdx.x;
  if (t >= N_EDGES * 8) return;
  int e = t >> 3, c = t & 7;
  atomicAdd(&agg[(size_t)edst[e] * 8 + c], (float)x[(size_t)esrc[e] * 8 + c] * g[e]);
}

// ---------------------------------------------------------------------------
// Node GEMM: Out(f16) = op( (A*scale) @ Bpack ), op = tanh optionally.
// One block per 16-row M-tile: A slab staged once into LDS as f16 (zero-padded
// to kchunks*32), then the 8 waves sweep the N-tiles; per K-chunk each lane
// does two conflict-free ds_load_b128 (row stride = kchunks*32+8 halves).
// ---------------------------------------------------------------------------
__global__ void wmma_gemm_kernel(const float* __restrict__ A, const _Float16* __restrict__ Bp,
                                 _Float16* __restrict__ Out, int K, int N,
                                 int ntiles, int kchunks, float scale, int do_tanh) {
  __shared__ alignas(16) _Float16 As[16 * 296];  // supports kchunks*32 <= 288
  int wave = threadIdx.x >> 5;
  int lane = threadIdx.x & 31;
  int mt = blockIdx.x;
  int kpad = kchunks * 32;
  int stride = kpad + 8;  // halves; 592B / 80B row pitch -> distinct LDS banks
  for (int idx = threadIdx.x; idx < 16 * kpad; idx += blockDim.x) {
    int row = idx / kpad, k = idx - row * kpad;
    float v = (k < K) ? A[(size_t)(mt * 16 + row) * K + k] : 0.f;
    As[row * stride + k] = (_Float16)v;
  }
  __syncthreads();
  const _Float16* arow = &As[(lane & 15) * stride];
  const int aoff = (lane & 16) ? 8 : 0;
  for (int nt = wave; nt < ntiles; nt += 8) {
    v8f acc = {};
    for (int kc = 0; kc < kchunks; ++kc) {
      HVec ua;
      ua.f[0] = *(const v4f*)(arow + kc * 32 + aoff);
      ua.f[1] = *(const v4f*)(arow + kc * 32 + 16 + aoff);
      v16h b = load_pack(Bp, nt * kchunks + kc, lane);
      acc = __builtin_amdgcn_wmma_f32_16x16x32_f16(false, ua.v, false, b, (short)0, acc,
                                                   false, false);
    }
    int col = nt * 16 + (lane & 15);
    if (col < N) {
#pragma unroll
      for (int r = 0; r < 8; ++r) {
        int row = mt * 16 + ((lane & 16) ? (8 + r) : r);
        float v = acc[r] * scale;
        if (do_tanh) v = tanhf(v);
        Out[(size_t)row * N + col] = (_Float16)v;
      }
    }
  }
}

// per-graph sum pool over contiguous 3000-node blocks, / sqrt(3000)
__global__ void pool_kernel(const _Float16* __restrict__ x3, float* __restrict__ out) {
  __shared__ float sm[3 * 256];
  int g = blockIdx.x;
  int t = threadIdx.x;
  float s0 = 0.f, s1 = 0.f, s2 = 0.f;
  for (int i = t; i < NPG; i += 256) {
    const _Float16* r = x3 + (size_t)(g * NPG + i) * 3;
    s0 += (float)r[0];
    s1 += (float)r[1];
    s2 += (float)r[2];
  }
  sm[t] = s0;
  sm[256 + t] = s1;
  sm[512 + t] = s2;
  __syncthreads();
  for (int off = 128; off; off >>= 1) {
    if (t < off) {
      sm[t] += sm[t + off];
      sm[256 + t] += sm[256 + t + off];
      sm[512 + t] += sm[512 + t + off];
    }
    __syncthreads();
  }
  if (t == 0) {
    float inv = rsqrtf((float)NPG);
    out[g * 3 + 0] = sm[0] * inv;
    out[g * 3 + 1] = sm[256] * inv;
    out[g * 3 + 2] = sm[512] * inv;
  }
}

// ---------------------------------------------------------------------------
extern "C" void kernel_launch(void* const* d_in, const int* in_sizes, int n_in,
                              void* d_out, int out_size, void* d_ws, size_t ws_size,
                              hipStream_t stream) {
  (void)in_sizes; (void)n_in; (void)out_size; (void)ws_size;
  const float* pos        = (const float*)d_in[0];
  const float* node_input = (const float*)d_in[1];
  const float* node_attr  = (const float*)d_in[2];
  const float* edge_attr  = (const float*)d_in[3];
  const int*   esrc       = (const int*)d_in[4];
  const int*   edst       = (const int*)d_in[5];
  const float* fc1        = (const float*)d_in[7];
  const float* b1         = (const float*)d_in[8];
  const float* fc2        = (const float*)d_in[9];
  const float* W[4] = {(const float*)d_in[10], (const float*)d_in[11],
                       (const float*)d_in[12], (const float*)d_in[13]};

  char* ws = (char*)d_ws;
  size_t off = 0;
  auto alloc = [&](size_t bytes) -> void* {
    void* p = ws + off;
    off += (bytes + 255) & ~(size_t)255;
    return p;
  };
  _Float16* xh    = (_Float16*)alloc((size_t)N_NODES * 288 * 2);  // activations (f16)
  float*    agg   = (float*)alloc((size_t)N_NODES * 288 * 4);     // f32 accumulator
  float*    ea    = (float*)alloc((size_t)N_EDGES * 10 * 4);
  _Float16* emb_h = (_Float16*)alloc((size_t)N_EDGES * 16 * 2);
  float*    g_all = (float*)alloc((size_t)4 * N_EDGES * 4);
  _Float16* fc1p  = (_Float16*)alloc((size_t)4 * 7 * 512 * 2);
  _Float16* fc2p  = (_Float16*)alloc((size_t)4 * 4 * 512 * 2);
  const int Kd[4] = {8, 288, 288, 288};
  const int Nd[4] = {288, 288, 288, 3};
  _Float16* Wp[4];
  for (int l = 0; l < 4; ++l) {
    int nt = (Nd[l] + 15) / 16, kc = (Kd[l] + 31) / 32;
    Wp[l] = (_Float16*)alloc((size_t)nt * kc * 512 * 2);
  }

  // 1) pack weights into WMMA B layout
  for (int l = 0; l < 4; ++l) {
    pack_b_kernel<<<(7 * 512 + 255) / 256, 256, 0, stream>>>(fc1 + l * 1000, fc1p + l * 7 * 512,
                                                             10, 100, 7, 1);
    pack_b_kernel<<<(4 * 512 + 255) / 256, 256, 0, stream>>>(fc2 + l * 1000, fc2p + l * 4 * 512,
                                                             100, 10, 1, 4);
  }
  for (int l = 0; l < 4; ++l) {
    int nt = (Nd[l] + 15) / 16, kc = (Kd[l] + 31) / 32;
    pack_b_kernel<<<(nt * kc * 512 + 255) / 256, 256, 0, stream>>>(W[l], Wp[l], Kd[l], Nd[l],
                                                                   nt, kc);
  }

  // 2) edge preprocessing
  edge_prep_kernel<<<(N_EDGES + 255) / 256, 256, 0, stream>>>(pos, edge_attr, esrc, edst, ea,
                                                              emb_h);
  // 3) x0
  x0_kernel<<<(N_NODES * 8 + 255) / 256, 256, 0, stream>>>(node_input, node_attr, xh);

  // 4) radial MLP gates for all 4 layers (WMMA)
  gate_kernel<<<N_EDGES / 16 / 8, 256, 0, stream>>>(emb_h, ea, fc1p, fc2p, b1, g_all);

  // 5) layers: scatter reads xh (f16), accumulates into agg (f32);
  //    GEMM reads agg, writes xh (f16) in place of the consumed activation.
  for (int l = 0; l < 4; ++l) {
    int din = Kd[l];
    int aggN = N_NODES * din;
    zero_kernel<<<(aggN + 255) / 256, 256, 0, stream>>>(agg, aggN);
    const float* g = g_all + (size_t)l * N_EDGES;
    if (din == 8)
      scatter8_kernel<<<(N_EDGES * 8 + 255) / 256, 256, 0, stream>>>(xh, g, esrc, edst, agg);
    else
      scatter288_kernel<<<15000, 256, 0, stream>>>(xh, g, esrc, edst, agg);
    int nt = (Nd[l] + 15) / 16, kc = (Kd[l] + 31) / 32;
    wmma_gemm_kernel<<<N_NODES / 16, 256, 0, stream>>>(agg, Wp[l], xh, Kd[l], Nd[l], nt, kc,
                                                       0.25f, l < 3 ? 1 : 0);
  }

  // 6) pool
  pool_kernel<<<N_GRAPHS, 256, 0, stream>>>(xh, (float*)d_out);
}
